// ChessRelativeTransformer_26886495273434
// MI455X (gfx1250) — compile-verified, matrix operand order checked by reading, not run on hardware
//
#include <hip/hip_runtime.h>
#include <hip/hip_bf16.h>

// ---------------------------------------------------------------------------
// Types for WMMA fragments (gfx1250, wave32)
// ---------------------------------------------------------------------------
typedef _Float16 h8  __attribute__((ext_vector_type(8)));
typedef _Float16 h16 __attribute__((ext_vector_type(16)));
typedef float    f8v __attribute__((ext_vector_type(8)));
typedef float    f4v __attribute__((ext_vector_type(4)));
typedef int      i4v __attribute__((ext_vector_type(4)));

__device__ __forceinline__ f8v wmma_f16(h16 a, h16 b, f8v c) {
  // D = A(16x32 f16) * B(32x16 f16) + C(16x16 f32)
  return __builtin_amdgcn_wmma_f32_16x16x32_f16(
      /*neg_a=*/false, a, /*neg_b=*/false, b,
      /*c_mod=*/(short)0, c, /*reuse_a=*/false, /*reuse_b=*/false);
}

// ---------------------------------------------------------------------------
// Async global->LDS copy (GLOBAL_LOAD_ASYNC_TO_LDS_B128, ASYNCcnt-tracked).
// Builtin signature (from clang diagnostic): (int4 AS1*, int4 AS3*, imm, imm).
// Falls back to a synchronous 16B copy when the builtin is unavailable.
// ---------------------------------------------------------------------------
#if defined(__gfx1250__) && __has_builtin(__builtin_amdgcn_global_load_async_to_lds_b128)
#define HAVE_ASYNC_LDS 1
#else
#define HAVE_ASYNC_LDS 0
#endif

__device__ __forceinline__ void cp16B(const _Float16* g, _Float16* l) {
#if HAVE_ASYNC_LDS
  __builtin_amdgcn_global_load_async_to_lds_b128(
      (__attribute__((address_space(1))) i4v*)g,
      (__attribute__((address_space(3))) i4v*)l, 0, 0);
#else
  *(h8*)l = *(const h8*)g;
#endif
}

__device__ __forceinline__ void cp_wait() {
#if HAVE_ASYNC_LDS
  asm volatile("s_wait_asynccnt 0x0" ::: "memory");
#endif
}

// NT fragment loader. Both A and B fragments of V_WMMA_F32_16X16X32_F16 read
// row (lane&15) of a row-major-along-K tile, in two contiguous 8-half runs:
//   k in [8g, 8g+8) and [8g+16, 8g+24),  g = lane>>4     (ISA 7.12.2 layout)
// => two ds_load_b128 per fragment.
__device__ __forceinline__ h16 frag_ld(const _Float16* base, int row0, int k0, int stride) {
  const int lane = threadIdx.x & 31;
  const int g = lane >> 4;
  const _Float16* p = base + (size_t)(row0 + (lane & 15)) * stride + k0 + 8 * g;
  h8 lo = *(const h8*)(p);
  h8 hi = *(const h8*)(p + 16);
  return __builtin_shufflevector(lo, hi, 0,1,2,3,4,5,6,7,8,9,10,11,12,13,14,15);
}

// ---------------------------------------------------------------------------
// Problem constants
// ---------------------------------------------------------------------------
#define BB       512
#define TT       64
#define DM       1024
#define NH       16
#define DH       64
#define NREL     225
#define RPADN    240   // NREL padded to WMMA N tiles (15 * 16)
#define RPADK    256   // NREL padded to WMMA K steps (8 * 32)
#define MTOT     (BB * TT)   // 32768

// ---------------------------------------------------------------------------
// Vectorized f32 -> f16 cast: 8 elements / thread (2x b128 load, 1x b128 store)
// ---------------------------------------------------------------------------
__global__ __launch_bounds__(256) void cast_f32_f16_v8(const float* __restrict__ src,
                                                       _Float16* __restrict__ dst, int n8) {
  int i = blockIdx.x * 256 + threadIdx.x;
  if (i < n8) {
    const f4v* s = (const f4v*)src;
    f4v a = s[2 * (size_t)i], b = s[2 * (size_t)i + 1];
    h8 o;
#pragma unroll
    for (int j = 0; j < 4; j++) { o[j] = (_Float16)a[j]; o[j + 4] = (_Float16)b[j]; }
    *(h8*)(dst + (size_t)i * 8) = o;
  }
}

// RR[h,r] = dot(rel_q[h,r,:], rel_k[h,r,:])
__global__ __launch_bounds__(256) void rr_kernel(const float* __restrict__ rq,
                                                 const float* __restrict__ rk,
                                                 float* __restrict__ RR) {
  int i = blockIdx.x * 256 + threadIdx.x;
  if (i < NH * NREL) {
    const float* a = rq + (size_t)i * DH;
    const float* b = rk + (size_t)i * DH;
    float s = 0.f;
#pragma unroll
    for (int d = 0; d < DH; d++) s += a[d] * b[d];
    RR[i] = s;
  }
}

// ---------------------------------------------------------------------------
// NT GEMM: C[m,n] = sum_k A[m,k] * W[n,k] + bias[n]
// A: (32768,1024) f16 row-major, W: (1024,1024) f16 row-major.
// OUT_MODE 0: f16, head-split (b,h,t,d) layout.   OUT_MODE 1: f32 row-major.
// 128x128 block tile, 256 threads = 8 waves, each wave a 32x64 sub-tile
// (2x4 WMMA tiles). K stepped by 32 through double-buffered LDS with async
// global->LDS copies pipelined one K-tile ahead.
// ---------------------------------------------------------------------------
template <int OUT_MODE>
__global__ __launch_bounds__(256) void gemm_proj(const _Float16* __restrict__ A,
                                                 const _Float16* __restrict__ W,
                                                 const float* __restrict__ bias,
                                                 _Float16* __restrict__ outH,
                                                 float* __restrict__ outF) {
  constexpr int K = DM, LDW = 40;       // 32 + 8 halves pad
  constexpr int TILEH = 128 * LDW;      // 5120 halves per buffer
  __shared__ _Float16 As[2 * TILEH];
  __shared__ _Float16 Ws[2 * TILEH];

  const int tid  = threadIdx.x;
  const int wave = tid >> 5, lane = tid & 31;
  const int m0 = blockIdx.y * 128, n0 = blockIdx.x * 128;
  const int wm0 = (wave & 3) * 32, wn0 = (wave >> 2) * 64;

  auto stage = [&](int k0, int bsel) {
#pragma unroll
    for (int c = tid; c < 512; c += 256) {       // 128 rows x 4 chunks of 8 halves
      const int row = c >> 2, col = (c & 3) * 8;
      cp16B(&A[(size_t)(m0 + row) * K + k0 + col], &As[bsel * TILEH + row * LDW + col]);
      cp16B(&W[(size_t)(n0 + row) * K + k0 + col], &Ws[bsel * TILEH + row * LDW + col]);
    }
  };

  f8v acc[2][4] = {};

  stage(0, 0);
  cp_wait();
  __syncthreads();

  int buf = 0;
  for (int k0 = 0; k0 < K; k0 += 32) {
    if (k0 + 32 < K) stage(k0 + 32, buf ^ 1);    // prefetch next tile into other buffer

    const _Float16* Ab = As + buf * TILEH;
    const _Float16* Wb = Ws + buf * TILEH;
    h16 af[2], bf[4];
#pragma unroll
    for (int i = 0; i < 2; i++) af[i] = frag_ld(Ab, wm0 + i * 16, 0, LDW);
#pragma unroll
    for (int j = 0; j < 4; j++) bf[j] = frag_ld(Wb, wn0 + j * 16, 0, LDW);
#pragma unroll
    for (int i = 0; i < 2; i++)
#pragma unroll
      for (int j = 0; j < 4; j++) acc[i][j] = wmma_f16(af[i], bf[j], acc[i][j]);

    cp_wait();
    __syncthreads();
    buf ^= 1;
  }

  const int g = lane >> 4, nl = lane & 15;
#pragma unroll
  for (int i = 0; i < 2; i++)
#pragma unroll
    for (int j = 0; j < 4; j++) {
      const int gn = n0 + wn0 + j * 16 + nl;
      const float bv = bias[gn];
#pragma unroll
      for (int r = 0; r < 8; r++) {
        const int gm = m0 + wm0 + i * 16 + g * 8 + r;  // C layout: M = r + 8*(lane>>4)
        const float v = acc[i][j][r] + bv;
        if (OUT_MODE == 0) {
          const int b = gm >> 6, t = gm & 63, h = gn >> 6, d = gn & 63;
          outH[(((size_t)(b * NH + h) * TT + t) * DH) + d] = (_Float16)v;
        } else {
          outF[(size_t)gm * DM + gn] = v;
        }
      }
    }
}

// ---------------------------------------------------------------------------
// Fused relative attention for one (b,h): 256 threads, dynamic LDS.
// LDS half-offsets:
//   qs   [64x72]      0      (reused as attn f16 after softmax)
//   ks   [64x72]      4608
//   vTs  [64x72]      9216   (v stored transposed [d][s])
//   rls  [17280]      13824  (rel_k rows -> rel_q rows -> rel_vT [64x264])
//   Qr   [64x240]     31104  \ overlaid later by P [64x264]
//   Kr   [64x240]     46464  /
//   Ls   f32 [64x64]  byte 123648
// total 140032 bytes dynamic shared (< 320 KB/WG on CDNA5).
// ---------------------------------------------------------------------------
__global__ __launch_bounds__(256) void attn_kernel(
    const _Float16* __restrict__ qg, const _Float16* __restrict__ kg,
    const _Float16* __restrict__ vg,                 // (B,H,T,DH) f16
    const float* __restrict__ rel_q, const float* __restrict__ rel_k,
    const float* __restrict__ rel_v,                 // (H,225,64) f32
    const int* __restrict__ rel_index,               // (64,64)
    const float* __restrict__ RR,                    // (H,225)
    _Float16* __restrict__ outO)                     // (B,T,1024) f16
{
  extern __shared__ char smem_raw[];
  _Float16* qs    = (_Float16*)smem_raw;
  _Float16* ks    = qs + 4608;
  _Float16* vTs   = ks + 4608;
  _Float16* rls   = vTs + 4608;
  _Float16* Qr    = rls + 17280;
  _Float16* Kr    = Qr + 64 * RPADN;
  _Float16* Pm    = Qr;        // overlay after gathers
  _Float16* attnb = qs;        // overlay after logits
  float*    Ls    = (float*)(smem_raw + 123648);

  const int h = blockIdx.x, b = blockIdx.y;
  const int tid = threadIdx.x, wave = tid >> 5, lane = tid & 31;
  const size_t bh = ((size_t)b * NH + h) * (TT * DH);

  // ---- stage q, k (async 16B copies), vT (transposed), rel_k rows ----
  for (int c = tid; c < 512; c += 256) {
    int row = c >> 3, col = (c & 7) * 8;
    cp16B(&qg[bh + row * DH + col], &qs[row * 72 + col]);
    cp16B(&kg[bh + row * DH + col], &ks[row * 72 + col]);
  }
  for (int e = tid; e < TT * DH; e += 256) {
    int d = e & 63, s = e >> 6;
    vTs[d * 72 + s] = vg[bh + (size_t)s * DH + d];
  }
  for (int e = tid; e < RPADN * DH; e += 256) {
    int d = e & 63, r = e >> 6;
    rls[r * 72 + d] = (r < NREL) ? (_Float16)rel_k[((size_t)h * NREL + r) * DH + d]
                                 : (_Float16)0.f;
  }
  cp_wait();
  __syncthreads();

  // ---- Qr = q @ rel_k^T  (64 x 240, K=64) ----
  for (int tIdx = wave; tIdx < 60; tIdx += 8) {
    int mt = tIdx / 15, nt = tIdx % 15;
    f8v acc = {};
#pragma unroll
    for (int k0 = 0; k0 < DH; k0 += 32)
      acc = wmma_f16(frag_ld(qs, mt * 16, k0, 72), frag_ld(rls, nt * 16, k0, 72), acc);
    const int n = nt * 16 + (lane & 15), g = lane >> 4;
#pragma unroll
    for (int r = 0; r < 8; r++) Qr[(mt * 16 + g * 8 + r) * RPADN + n] = (_Float16)acc[r];
  }
  __syncthreads();

  // ---- restage rel_q, compute Kr = k @ rel_q^T ----
  for (int e = tid; e < RPADN * DH; e += 256) {
    int d = e & 63, r = e >> 6;
    rls[r * 72 + d] = (r < NREL) ? (_Float16)rel_q[((size_t)h * NREL + r) * DH + d]
                                 : (_Float16)0.f;
  }
  __syncthreads();
  for (int tIdx = wave; tIdx < 60; tIdx += 8) {
    int mt = tIdx / 15, nt = tIdx % 15;
    f8v acc = {};
#pragma unroll
    for (int k0 = 0; k0 < DH; k0 += 32)
      acc = wmma_f16(frag_ld(ks, mt * 16, k0, 72), frag_ld(rls, nt * 16, k0, 72), acc);
    const int n = nt * 16 + (lane & 15), g = lane >> 4;
#pragma unroll
    for (int r = 0; r < 8; r++) Kr[(mt * 16 + g * 8 + r) * RPADN + n] = (_Float16)acc[r];
  }
  __syncthreads();

  // ---- stage rel_vT [d][264] while computing logits = q k^T + gathers ----
  for (int e = tid; e < DH * 264; e += 256) {
    int r = e % 264, d = e / 264;
    rls[d * 264 + r] = (r < NREL) ? (_Float16)rel_v[((size_t)h * NREL + r) * DH + d]
                                  : (_Float16)0.f;
  }
  for (int tIdx = wave; tIdx < 16; tIdx += 8) {
    int mt = tIdx >> 2, nt = tIdx & 3;
    f8v acc = {};
#pragma unroll
    for (int k0 = 0; k0 < DH; k0 += 32)
      acc = wmma_f16(frag_ld(qs, mt * 16, k0, 72), frag_ld(ks, nt * 16, k0, 72), acc);
    const int s = nt * 16 + (lane & 15), g = lane >> 4;
#pragma unroll
    for (int r = 0; r < 8; r++) {
      const int t = mt * 16 + g * 8 + r;
      const int idx = rel_index[t * TT + s];
      Ls[t * TT + s] = acc[r] + (float)Qr[t * RPADN + idx] +
                       (float)Kr[s * RPADN + idx] + RR[h * NREL + idx];
    }
  }
  __syncthreads();

  // ---- zero P (overlays Qr/Kr) + softmax (one row per thread) ----
  for (int e = tid; e < 64 * 264; e += 256) Pm[e] = (_Float16)0.f;
  if (tid < TT) {
    const int t = tid;
    const float scale = 0.125f;  // 1/sqrt(64)
    float mx = -1e30f;
    for (int s = 0; s < TT; s++) mx = fmaxf(mx, Ls[t * TT + s]);
    float sum = 0.f;
    for (int s = 0; s < TT; s++) sum += __expf((Ls[t * TT + s] - mx) * scale);
    const float inv = 1.f / sum;
    for (int s = 0; s < TT; s++)
      attnb[t * 72 + s] = (_Float16)(__expf((Ls[t * TT + s] - mx) * scale) * inv);
  }
  __syncthreads();

  // ---- scatter P[t,r] += attn[t,s] (row t owned by one thread: no races) ----
  if (tid < TT) {
    const int t = tid;
    for (int s = 0; s < TT; s++) {
      const int idx = rel_index[t * TT + s];
      Pm[t * 264 + idx] = (_Float16)((float)Pm[t * 264 + idx] + (float)attnb[t * 72 + s]);
    }
  }
  __syncthreads();

  // ---- out = attn @ v  +  P @ rel_v  (both WMMA) ----
  for (int tIdx = wave; tIdx < 16; tIdx += 8) {
    int mt = tIdx >> 2, nt = tIdx & 3;
    f8v acc = {};
#pragma unroll
    for (int k0 = 0; k0 < TT; k0 += 32)
      acc = wmma_f16(frag_ld(attnb, mt * 16, k0, 72), frag_ld(vTs, nt * 16, k0, 72), acc);
#pragma unroll
    for (int k0 = 0; k0 < RPADK; k0 += 32)
      acc = wmma_f16(frag_ld(Pm, mt * 16, k0, 264), frag_ld(rls, nt * 16, k0, 264), acc);
    const int g = lane >> 4, d = nt * 16 + (lane & 15);
#pragma unroll
    for (int r = 0; r < 8; r++) {
      const int t = mt * 16 + g * 8 + r;
      outO[((size_t)b * TT + t) * DM + h * DH + d] = (_Float16)acc[r];
    }
  }
}

// ---------------------------------------------------------------------------
// Host launcher
// ---------------------------------------------------------------------------
extern "C" void kernel_launch(void* const* d_in, const int* in_sizes, int n_in,
                              void* d_out, int out_size, void* d_ws, size_t ws_size,
                              hipStream_t stream) {
  (void)in_sizes; (void)n_in; (void)out_size; (void)ws_size;
  const float* x     = (const float*)d_in[0];
  const float* Wq_w  = (const float*)d_in[1];
  const float* Wq_b  = (const float*)d_in[2];
  const float* Wk_w  = (const float*)d_in[3];
  const float* Wk_b  = (const float*)d_in[4];
  const float* Wv_w  = (const float*)d_in[5];
  const float* Wv_b  = (const float*)d_in[6];
  const float* Wo_w  = (const float*)d_in[7];
  const float* Wo_b  = (const float*)d_in[8];
  const float* rel_q = (const float*)d_in[9];
  const float* rel_k = (const float*)d_in[10];
  const float* rel_v = (const float*)d_in[11];
  const int*   ridx  = (const int*)d_in[12];

  char* ws = (char*)d_ws;
  // workspace layout (bytes)
  _Float16* xh = (_Float16*)(ws);                       // 67,108,864  (x f16; reused as oh)
  _Float16* wh = (_Float16*)(ws + 67108864);            //  8,388,608  (Wq,Wk,Wv,Wo f16)
  _Float16* qh = (_Float16*)(ws + 75497472);            // 67,108,864  (B,H,T,64)
  _Float16* kh = (_Float16*)(ws + 142606336);           // 67,108,864
  _Float16* vh = (_Float16*)(ws + 209715200);           // 67,108,864
  float*    RR = (float*)(ws + 276824064);              //     14,400
  _Float16* oh = xh;  // x consumed before attention writes here (same stream)

  const int nx = MTOT * DM;     // 33,554,432
  const int nw = DM * DM;       //  1,048,576
  cast_f32_f16_v8<<<nx / 8 / 256, 256, 0, stream>>>(x,    xh,                  nx / 8);
  cast_f32_f16_v8<<<nw / 8 / 256, 256, 0, stream>>>(Wq_w, wh + 0 * (size_t)nw, nw / 8);
  cast_f32_f16_v8<<<nw / 8 / 256, 256, 0, stream>>>(Wk_w, wh + 1 * (size_t)nw, nw / 8);
  cast_f32_f16_v8<<<nw / 8 / 256, 256, 0, stream>>>(Wv_w, wh + 2 * (size_t)nw, nw / 8);
  cast_f32_f16_v8<<<nw / 8 / 256, 256, 0, stream>>>(Wo_w, wh + 3 * (size_t)nw, nw / 8);
  rr_kernel<<<(NH * NREL + 255) / 256, 256, 0, stream>>>(rel_q, rel_k, RR);

  dim3 gproj(DM / 128, MTOT / 128);  // (8, 256)
  gemm_proj<0><<<gproj, 256, 0, stream>>>(xh, wh + 0 * (size_t)nw, Wq_b, qh, nullptr);
  gemm_proj<0><<<gproj, 256, 0, stream>>>(xh, wh + 1 * (size_t)nw, Wk_b, kh, nullptr);
  gemm_proj<0><<<gproj, 256, 0, stream>>>(xh, wh + 2 * (size_t)nw, Wv_b, vh, nullptr);

  attn_kernel<<<dim3(NH, BB), 256, 140032, stream>>>(qh, kh, vh, rel_q, rel_k, rel_v,
                                                     ridx, RR, oh);

  gemm_proj<1><<<gproj, 256, 0, stream>>>(oh, wh + 3 * (size_t)nw, Wo_b, nullptr,
                                          (float*)d_out);
}
